// MambaCDCBlock_68487548502481
// MI455X (gfx1250) — matrix operand order repaired and checked by reading, not compiled, not run on hardware
//
#include <hip/hip_runtime.h>
#include <hip/hip_bf16.h>
#include <stdint.h>

// ---------------------------------------------------------------------------
// MambaCDCBlock for MI455X (gfx1250, wave32, WMMA).
// All 4 dense GEMMs run on v_wmma_f32_16x16x32_f16 (f16 in, f32 accum).
// Weight tile staged once per block into LDS (padded, conflict-free
// ds_load_b128 fragment reads); A fragments register double-buffered so
// global b128 clauses overlap WMMA issue. Epilogues use compile-time strides
// (immediate-offset store clauses) and b128 vector stores where rows are
// memory-contiguous. Selective scan is a 3-pass chunked scan.
// ---------------------------------------------------------------------------

typedef __attribute__((ext_vector_type(16))) _Float16     v16h;
typedef __attribute__((ext_vector_type(8)))  float        v8f;
typedef __attribute__((ext_vector_type(4)))  float        v4f;
typedef __attribute__((ext_vector_type(4)))  unsigned int v4u;

#define BATCH   4
#define CIN     256
#define COUT    256
#define HH      64
#define WWID    64
#define LL      4096
#define PCHUNK  4
#define DMDIM   64
#define DSTATE  16
#define DCONV   4
#define DINNER  128
#define DTRANK  4
#define BMB     16      // PCHUNK * BATCH
#define NCHUNK  64
#define CHLEN   64      // LL / NCHUNK

union HFrag { v16h v; v4u q[2]; };

// ---------------- f32 -> f16 weight conversion ------------------------------
__global__ void k_cvt_f16(const float* __restrict__ src, _Float16* __restrict__ dst, int n) {
  int i = blockIdx.x * blockDim.x + threadIdx.x;
  if (i < n) dst[i] = (_Float16)src[i];
}

// ---------------- LayerNorm #1 + channel-chunk split ------------------------
// x [B, C, L] -> xh f16 / xcf f32 laid out [bm = p*B+b][l][dm] (row-major K=64)
__global__ void k_ln1(const float* __restrict__ x, const float* __restrict__ g,
                      const float* __restrict__ bta, _Float16* __restrict__ xh,
                      float* __restrict__ xcf) {
  int tid = blockIdx.x * blockDim.x + threadIdx.x;
  if (tid >= BATCH * LL) return;
  int b = tid >> 12, l = tid & (LL - 1);
  const float* xp = x + (size_t)b * CIN * LL + l;
  float s = 0.f;
  for (int c = 0; c < CIN; ++c) s += xp[(size_t)c * LL];
  float mu = s * (1.f / CIN);
  float v = 0.f;
  for (int c = 0; c < CIN; ++c) { float d = xp[(size_t)c * LL] - mu; v += d * d; }
  float rs = rsqrtf(v * (1.f / CIN) + 1e-5f);
  for (int c = 0; c < CIN; ++c) {
    float val = (xp[(size_t)c * LL] - mu) * rs * g[c] + bta[c];
    int p = c >> 6, dm = c & 63;
    size_t o = ((size_t)(p * BATCH + b) * LL + l) * DMDIM + dm;
    xh[o]  = (_Float16)val;
    xcf[o] = val;
  }
}

// ---------------- WMMA GEMM: D = A(f16,[M,K]) x Bw(f16,[N,K])^T -------------
// Weight tile (64 x K) staged in LDS once per block; A double-buffered.
// MODE 0: dstF[m*N+n]                                   (in_proj -> xz)
// MODE 1: skip-add + scatter into ymid [B,L,256]        (out_proj)
// MODE 2: +bias, NCHW store into outbuf [B,C,L]         (proj)
// MODE 3: +outbuf residual, store d_out [B,C,L]         (pw conv)
template <int MODE, int K, int N>
__global__ void k_gemm_wmma(const _Float16* __restrict__ A, const _Float16* __restrict__ Bw,
                            float* __restrict__ dstF, const float* __restrict__ auxA,
                            const float* __restrict__ auxB, int M) {
  constexpr int KP = K + 8;                 // padded row stride (halves): bank-quad skew
  __shared__ _Float16 sB[64 * KP];

  int tid  = threadIdx.x;
  int wave = tid >> 5;
  int lane = tid & 31;
  int l16  = lane & 15;
  int h16  = lane >> 4;
  int row0 = blockIdx.x * 128 + wave * 16;
  int col0 = blockIdx.y * 64;

  // cooperative stage of the 64 x K weight tile (shared by all 8 waves)
  constexpr int NCH = 64 * K / 8;           // number of 8-half chunks
  for (int cid = tid; cid < NCH; cid += 256) {
    int row = cid / (K / 8);
    int c8  = cid % (K / 8);
    *(v4u*)(sB + row * KP + c8 * 8) =
        *(const v4u*)(Bw + (size_t)(col0 + row) * K + c8 * 8);
  }
  __syncthreads();

  v8f acc[4] = {};
  const _Float16* ap = A + (size_t)(row0 + l16) * K + h16 * 8;

  HFrag a_cur;
  a_cur.q[0] = *(const v4u*)(ap);
  a_cur.q[1] = *(const v4u*)(ap + 16);

#pragma unroll
  for (int k0 = 0; k0 < K; k0 += 32) {
    HFrag a_nxt;
    if (k0 + 32 < K) {
      a_nxt.q[0] = *(const v4u*)(ap + k0 + 32);
      a_nxt.q[1] = *(const v4u*)(ap + k0 + 48);
      __builtin_prefetch(ap + k0 + 64, 0, 1);   // global_prefetch_b8
    }
#pragma unroll
    for (int nt = 0; nt < 4; ++nt) {
      const _Float16* bp = sB + (nt * 16 + l16) * KP + k0 + h16 * 8;
      HFrag bf;
      bf.q[0] = *(const v4u*)(bp);              // ds_load_b128, conflict-free
      bf.q[1] = *(const v4u*)(bp + 16);
      acc[nt] = __builtin_amdgcn_wmma_f32_16x16x32_f16(
          false, a_cur.v, false, bf.v, (short)0, acc[nt], false, false);
    }
    if (k0 + 32 < K) a_cur = a_nxt;
  }

  // -------- epilogue: compile-time strides -> immediate-offset stores -------
  int m0 = row0 + h16 * 8;                    // 8 consecutive rows per lane
#pragma unroll
  for (int nt = 0; nt < 4; ++nt) {
    int n = col0 + nt * 16 + l16;
    if (MODE == 0) {
      float* dp = dstF + (size_t)m0 * N + n;
#pragma unroll
      for (int r = 0; r < 8; ++r) dp[r * N] = acc[nt][r];
    } else if (MODE == 1) {
      // m0..m0+7 share the same bm (tiles never straddle an L boundary)
      int bm = m0 >> 12, l0 = m0 & (LL - 1);
      int p = bm >> 2, b = bm & 3;
      float sk = auxB[0];
      const float* xp = auxA + (size_t)m0 * DMDIM + n;
      float* dp = dstF + ((size_t)b * LL + l0) * CIN + p * DMDIM + n;
#pragma unroll
      for (int r = 0; r < 8; ++r) dp[r * CIN] = acc[nt][r] + sk * xp[r * DMDIM];
    } else if (MODE == 2) {
      int b = m0 >> 12, l0 = m0 & (LL - 1);   // rows contiguous in l
      float bias = auxA[n];
      float* dp = dstF + ((size_t)b * COUT + n) * LL + l0;
      v4f lo = {acc[nt][0] + bias, acc[nt][1] + bias, acc[nt][2] + bias, acc[nt][3] + bias};
      v4f hi = {acc[nt][4] + bias, acc[nt][5] + bias, acc[nt][6] + bias, acc[nt][7] + bias};
      *(v4f*)dp = lo;                         // global_store_b128
      *(v4f*)(dp + 4) = hi;
    } else {
      int b = m0 >> 12, l0 = m0 & (LL - 1);
      size_t base = ((size_t)b * COUT + n) * LL + l0;
      v4f o0 = *(const v4f*)(auxA + base);    // residual, global_load_b128
      v4f o1 = *(const v4f*)(auxA + base + 4);
      v4f lo = {acc[nt][0] + o0[0], acc[nt][1] + o0[1], acc[nt][2] + o0[2], acc[nt][3] + o0[3]};
      v4f hi = {acc[nt][4] + o1[0], acc[nt][5] + o1[1], acc[nt][6] + o1[2], acc[nt][7] + o1[3]};
      *(v4f*)(dstF + base) = lo;
      *(v4f*)(dstF + base + 4) = hi;
    }
  }
}

// ---------------- causal depthwise conv1d (D_CONV=4) + SiLU -----------------
__global__ void k_conv_silu(const float* __restrict__ xz, const float* __restrict__ cw,
                            const float* __restrict__ cb, float* __restrict__ xs,
                            _Float16* __restrict__ xsh) {
  size_t tid = (size_t)blockIdx.x * blockDim.x + threadIdx.x;
  if (tid >= (size_t)BMB * LL * DINNER) return;
  int d  = (int)(tid & (DINNER - 1));
  int t  = (int)((tid >> 7) & (LL - 1));
  int bm = (int)(tid >> 19);
  const float* xp = xz + (size_t)bm * LL * (2 * DINNER) + d;
  float s = cb[d];
#pragma unroll
  for (int j = 0; j < DCONV; ++j) {
    int tt = t - (DCONV - 1) + j;
    if (tt >= 0) s += cw[d * DCONV + j] * xp[(size_t)tt * (2 * DINNER)];
  }
  float y = s / (1.f + expf(-s));
  xs[tid]  = y;
  xsh[tid] = (_Float16)y;
}

// ---------------- x_proj: [Bm*L,128] x [36,128]^T -> dt/B/C -----------------
__global__ void k_xproj(const float* __restrict__ xs, const float* __restrict__ xw,
                        float* __restrict__ dtb, float* __restrict__ Bm_,
                        float* __restrict__ Cm_) {
  size_t tid = (size_t)blockIdx.x * blockDim.x + threadIdx.x;
  if (tid >= (size_t)BMB * LL * 36) return;
  int n = (int)(tid % 36);
  size_t r = tid / 36;
  const float* xp = xs + r * DINNER;
  const float* wp = xw + n * DINNER;
  float s = 0.f;
  for (int k = 0; k < DINNER; ++k) s += xp[k] * wp[k];
  if (n < DTRANK)               dtb[r * DTRANK + n] = s;
  else if (n < DTRANK + DSTATE) Bm_[r * DSTATE + (n - DTRANK)] = s;
  else                          Cm_[r * DSTATE + (n - DTRANK - DSTATE)] = s;
}

// ---------------- delta = softplus(dt x dt_proj^T + bias) -------------------
__global__ void k_delta(const float* __restrict__ dtb, const float* __restrict__ dtw,
                        const float* __restrict__ dtbias, float* __restrict__ delta) {
  size_t tid = (size_t)blockIdx.x * blockDim.x + threadIdx.x;
  if (tid >= (size_t)BMB * LL * DINNER) return;
  int d = (int)(tid & (DINNER - 1));
  size_t r = tid >> 7;
  float s = dtbias[d];
#pragma unroll
  for (int j = 0; j < DTRANK; ++j) s += dtb[r * DTRANK + j] * dtw[d * DTRANK + j];
  delta[tid] = (s > 20.f) ? s : log1pf(expf(s));
}

// ---------------- scan pass 1: per-chunk decay product + local final --------
__global__ void k_scan1(const float* __restrict__ delta, const float* __restrict__ xs,
                        const float* __restrict__ Bm_, const float* __restrict__ Alog,
                        float* __restrict__ chP, float* __restrict__ chF) {
  int bm = blockIdx.x, ch = blockIdx.y, d = threadIdx.x;
  float Ad[DSTATE], Pp[DSTATE], Ff[DSTATE];
#pragma unroll
  for (int s = 0; s < DSTATE; ++s) { Ad[s] = -expf(Alog[d * DSTATE + s]); Pp[s] = 1.f; Ff[s] = 0.f; }
  size_t base = (size_t)bm * LL + (size_t)ch * CHLEN;
  for (int t = 0; t < CHLEN; ++t) {
    size_t r = base + t;
    float dl = delta[r * DINNER + d];
    float u  = xs[r * DINNER + d];
    float du = dl * u;
    const float* bp = Bm_ + r * DSTATE;
#pragma unroll
    for (int s = 0; s < DSTATE; ++s) {
      float a = __expf(dl * Ad[s]);
      Pp[s] *= a;
      Ff[s] = Ff[s] * a + du * bp[s];
    }
  }
  size_t o = (((size_t)bm * NCHUNK + ch) * DINNER + d) * DSTATE;
#pragma unroll
  for (int s = 0; s < DSTATE; ++s) { chP[o + s] = Pp[s]; chF[o + s] = Ff[s]; }
}

// ---------------- scan combine: sequential prefix over 64 chunks ------------
__global__ void k_scan_combine(const float* __restrict__ chP, const float* __restrict__ chF,
                               float* __restrict__ chH) {
  int tid = blockIdx.x * blockDim.x + threadIdx.x;
  if (tid >= BMB * DINNER * DSTATE) return;
  int s  = tid & 15;
  int d  = (tid >> 4) & (DINNER - 1);
  int bm = tid >> 11;
  float h = 0.f;
  for (int ch = 0; ch < NCHUNK; ++ch) {
    size_t o = (((size_t)bm * NCHUNK + ch) * DINNER + d) * DSTATE + s;
    chH[o] = h;
    h = chF[o] + chP[o] * h;
  }
}

// ---------------- scan pass 2: replay with init state, fuse y epilogue ------
__global__ void k_scan2(const float* __restrict__ delta, const float* __restrict__ xs,
                        const float* __restrict__ Bm_, const float* __restrict__ Cm_,
                        const float* __restrict__ Alog, const float* __restrict__ Dvec,
                        const float* __restrict__ xz, const float* __restrict__ chH,
                        _Float16* __restrict__ yh) {
  int bm = blockIdx.x, ch = blockIdx.y, d = threadIdx.x;
  float Ad[DSTATE], h[DSTATE];
  size_t o = (((size_t)bm * NCHUNK + ch) * DINNER + d) * DSTATE;
#pragma unroll
  for (int s = 0; s < DSTATE; ++s) { Ad[s] = -expf(Alog[d * DSTATE + s]); h[s] = chH[o + s]; }
  float Dd = Dvec[d];
  size_t base = (size_t)bm * LL + (size_t)ch * CHLEN;
  for (int t = 0; t < CHLEN; ++t) {
    size_t r = base + t;
    float dl = delta[r * DINNER + d];
    float u  = xs[r * DINNER + d];
    float du = dl * u;
    const float* bp = Bm_ + r * DSTATE;
    const float* cp = Cm_ + r * DSTATE;
    float y = 0.f;
#pragma unroll
    for (int s = 0; s < DSTATE; ++s) {
      float a = __expf(dl * Ad[s]);
      h[s] = h[s] * a + du * bp[s];
      y += h[s] * cp[s];
    }
    y += u * Dd;
    float z = xz[r * (2 * DINNER) + DINNER + d];
    y *= z / (1.f + expf(-z));
    yh[r * DINNER + d] = (_Float16)y;
  }
}

// ---------------- LayerNorm #2 over [B*L, 256] rows -------------------------
__global__ void k_ln2(const float* __restrict__ ymid, const float* __restrict__ g,
                      const float* __restrict__ bta, _Float16* __restrict__ ymh) {
  int tid = blockIdx.x * blockDim.x + threadIdx.x;
  if (tid >= BATCH * LL) return;
  const float* yp = ymid + (size_t)tid * CIN;
  float s = 0.f;
  for (int c = 0; c < CIN; ++c) s += yp[c];
  float mu = s * (1.f / CIN);
  float v = 0.f;
  for (int c = 0; c < CIN; ++c) { float d = yp[c] - mu; v += d * d; }
  float rs = rsqrtf(v * (1.f / CIN) + 1e-5f);
  for (int c = 0; c < CIN; ++c)
    ymh[(size_t)tid * CIN + c] = (_Float16)((yp[c] - mu) * rs * g[c] + bta[c]);
}

// ---------------- theta = sigmoid(sum_c out*theta_w + theta_b) --------------
__global__ void k_theta(const float* __restrict__ outbuf, const float* __restrict__ tw,
                        const float* __restrict__ tb, float* __restrict__ th) {
  int tid = blockIdx.x * blockDim.x + threadIdx.x;
  if (tid >= BATCH * LL) return;
  int b = tid >> 12, l = tid & (LL - 1);
  float s = tb[0];
  for (int c = 0; c < COUT; ++c) s += outbuf[((size_t)b * COUT + c) * LL + l] * tw[c];
  th[tid] = 1.f / (1.f + expf(-s));
}

// ---------------- CDC: dw3x3 + central-diff + exact GELU -> gh [B*L,C] f16 --
__global__ void k_cdc(const float* __restrict__ outbuf, const float* __restrict__ cdcw,
                      const float* __restrict__ th, _Float16* __restrict__ gh) {
  size_t tid = (size_t)blockIdx.x * blockDim.x + threadIdx.x;
  if (tid >= (size_t)BATCH * COUT * LL) return;
  int hw = (int)(tid & (LL - 1));
  int c  = (int)((tid >> 12) & (COUT - 1));
  int b  = (int)(tid >> 20);
  int h = hw >> 6, w = hw & (WWID - 1);
  const float* op = outbuf + ((size_t)b * COUT + c) * LL;
  const float* wp = cdcw + c * 9;
  float ks = 0.f, cv = 0.f;
#pragma unroll
  for (int dy = 0; dy < 3; ++dy)
#pragma unroll
    for (int dx = 0; dx < 3; ++dx) {
      float wgt = wp[dy * 3 + dx];
      ks += wgt;
      int hh = h + dy - 1, ww = w + dx - 1;
      if (hh >= 0 && hh < HH && ww >= 0 && ww < WWID) cv += wgt * op[hh * WWID + ww];
    }
  float edge = cv - op[hw] * ks;
  float cd = cv + th[b * LL + hw] * edge;
  float ge = 0.5f * cd * (1.f + erff(cd * 0.70710678118654752f));
  gh[((size_t)b * LL + hw) * COUT + c] = (_Float16)ge;
}

// ---------------------------------------------------------------------------
extern "C" void kernel_launch(void* const* d_in, const int* in_sizes, int n_in,
                              void* d_out, int out_size, void* d_ws, size_t ws_size,
                              hipStream_t stream) {
  const float* x         = (const float*)d_in[0];
  const float* ln_g      = (const float*)d_in[1];
  const float* ln_b      = (const float*)d_in[2];
  const float* in_proj_w = (const float*)d_in[3];
  const float* conv_w    = (const float*)d_in[4];
  const float* conv_b    = (const float*)d_in[5];
  const float* x_proj_w  = (const float*)d_in[6];
  const float* dt_proj_w = (const float*)d_in[7];
  const float* dt_proj_b = (const float*)d_in[8];
  const float* A_log     = (const float*)d_in[9];
  const float* Dvec      = (const float*)d_in[10];
  const float* out_proj_w= (const float*)d_in[11];
  const float* skip      = (const float*)d_in[12];
  const float* proj_w    = (const float*)d_in[13];
  const float* proj_b    = (const float*)d_in[14];
  const float* cdc_w     = (const float*)d_in[15];
  const float* theta_w   = (const float*)d_in[16];
  const float* theta_b   = (const float*)d_in[17];
  const float* pw_w      = (const float*)d_in[18];
  float* out = (float*)d_out;

  size_t off = 0;
  auto alloc = [&](size_t bytes) -> void* {
    void* p = (void*)((char*)d_ws + off);
    off += (bytes + 255) & ~(size_t)255;
    return p;
  };
  const size_t ML = (size_t)BMB * LL;     // 65536
  const size_t BL = (size_t)BATCH * LL;   // 16384

  _Float16* w_inproj  = (_Float16*)alloc((size_t)2 * DINNER * DMDIM * 2);
  _Float16* w_outproj = (_Float16*)alloc((size_t)DMDIM * DINNER * 2);
  _Float16* w_proj    = (_Float16*)alloc((size_t)COUT * CIN * 2);
  _Float16* w_pw      = (_Float16*)alloc((size_t)COUT * COUT * 2);
  _Float16* xh        = (_Float16*)alloc(ML * DMDIM * 2);
  float*    xcf       = (float*)   alloc(ML * DMDIM * 4);
  float*    xz        = (float*)   alloc(ML * 2 * DINNER * 4);
  float*    xs        = (float*)   alloc(ML * DINNER * 4);
  _Float16* xsh       = (_Float16*)alloc(ML * DINNER * 2);
  float*    dtb       = (float*)   alloc(ML * DTRANK * 4);
  float*    Bm_       = (float*)   alloc(ML * DSTATE * 4);
  float*    Cm_       = (float*)   alloc(ML * DSTATE * 4);
  float*    delta     = (float*)   alloc(ML * DINNER * 4);
  float*    chP       = (float*)   alloc((size_t)BMB * NCHUNK * DINNER * DSTATE * 4);
  float*    chF       = (float*)   alloc((size_t)BMB * NCHUNK * DINNER * DSTATE * 4);
  float*    chH       = (float*)   alloc((size_t)BMB * NCHUNK * DINNER * DSTATE * 4);
  _Float16* yh        = (_Float16*)alloc(ML * DINNER * 2);
  float*    ymid      = (float*)   alloc(BL * CIN * 4);
  _Float16* ymh       = (_Float16*)alloc(BL * CIN * 2);
  float*    outbuf    = (float*)   alloc(BL * COUT * 4);
  float*    th        = (float*)   alloc(BL * 4);
  _Float16* gh        = (_Float16*)alloc(BL * COUT * 2);

  // weight conversions
  k_cvt_f16<<<(2 * DINNER * DMDIM + 255) / 256, 256, 0, stream>>>(in_proj_w, w_inproj, 2 * DINNER * DMDIM);
  k_cvt_f16<<<(DMDIM * DINNER + 255) / 256, 256, 0, stream>>>(out_proj_w, w_outproj, DMDIM * DINNER);
  k_cvt_f16<<<(COUT * CIN + 255) / 256, 256, 0, stream>>>(proj_w, w_proj, COUT * CIN);
  k_cvt_f16<<<(COUT * COUT + 255) / 256, 256, 0, stream>>>(pw_w, w_pw, COUT * COUT);

  // LN1 + split
  k_ln1<<<(int)((BL + 255) / 256), 256, 0, stream>>>(x, ln_g, ln_b, xh, xcf);

  // in_proj GEMM: [65536,64] x [256,64]^T -> xz
  k_gemm_wmma<0, DMDIM, 2 * DINNER><<<dim3((unsigned)(ML / 128), (2 * DINNER) / 64), 256, 0, stream>>>(
      xh, w_inproj, xz, nullptr, nullptr, (int)ML);

  // causal conv + silu
  k_conv_silu<<<(int)((ML * DINNER + 255) / 256), 256, 0, stream>>>(xz, conv_w, conv_b, xs, xsh);

  // x_proj + delta
  k_xproj<<<(int)((ML * 36 + 255) / 256), 256, 0, stream>>>(xs, x_proj_w, dtb, Bm_, Cm_);
  k_delta<<<(int)((ML * DINNER + 255) / 256), 256, 0, stream>>>(dtb, dt_proj_w, dt_proj_b, delta);

  // chunked parallel scan
  k_scan1<<<dim3(BMB, NCHUNK), DINNER, 0, stream>>>(delta, xs, Bm_, A_log, chP, chF);
  k_scan_combine<<<(BMB * DINNER * DSTATE) / 256, 256, 0, stream>>>(chP, chF, chH);
  k_scan2<<<dim3(BMB, NCHUNK), DINNER, 0, stream>>>(delta, xs, Bm_, Cm_, A_log, Dvec, xz, chH, yh);

  // out_proj GEMM + skip + scatter to [B,L,256]
  k_gemm_wmma<1, DINNER, DMDIM><<<dim3((unsigned)(ML / 128), 1), 256, 0, stream>>>(
      yh, w_outproj, ymid, xcf, skip, (int)ML);

  // LN2
  k_ln2<<<(int)((BL + 255) / 256), 256, 0, stream>>>(ymid, ln_g, ln_b, ymh);

  // proj GEMM (+bias) -> outbuf [B,C,L]
  k_gemm_wmma<2, CIN, COUT><<<dim3((unsigned)(BL / 128), COUT / 64), 256, 0, stream>>>(
      ymh, w_proj, outbuf, proj_b, nullptr, (int)BL);

  // theta + CDC + GELU
  k_theta<<<(int)((BL + 255) / 256), 256, 0, stream>>>(outbuf, theta_w, theta_b, th);
  k_cdc<<<(int)((BL * COUT + 255) / 256), 256, 0, stream>>>(outbuf, cdc_w, th, gh);

  // pw GEMM + residual -> d_out [B,C,H,W]
  k_gemm_wmma<3, COUT, COUT><<<dim3((unsigned)(BL / 128), COUT / 64), 256, 0, stream>>>(
      gh, w_pw, out, outbuf, nullptr, (int)BL);
}
